// SelfAttention_71390946394391
// MI455X (gfx1250) — matrix operand order repaired and checked by reading, not compiled, not run on hardware
//
#include <hip/hip_runtime.h>
#include <hip/hip_bf16.h>

#define BB   8
#define CH   256
#define NPIX 4096
#define DQK  32

typedef __attribute__((ext_vector_type(16))) __bf16       v16bf;
typedef __attribute__((ext_vector_type(8)))  float        v8f;
typedef __attribute__((ext_vector_type(4)))  unsigned int u32x4;
typedef __attribute__((ext_vector_type(8)))  int          i32x8;
typedef __attribute__((ext_vector_type(4)))  int          i32x4;

union FragU { v16bf m; u32x4 u[2]; };

// A-fragment (16x32 bf16, M=lane%16). rowChunk -> start of this lane's 32-element K chunk.
// lanes 0-15: K [0..7],[16..23] ; lanes 16-31: K [8..15],[24..31]
__device__ __forceinline__ v16bf load_fragA(const __hip_bfloat16* rowChunk, int lane) {
  FragU f;
  const int off = (lane < 16) ? 0 : 8;
  f.u[0] = *(const u32x4*)(rowChunk + off);
  f.u[1] = *(const u32x4*)(rowChunk + 16 + off);
  return f.m;
}
// B-fragment (32x16 bf16, col=lane%16). colK0 -> K=0 element of this lane's column.
__device__ __forceinline__ v16bf load_fragB(const __hip_bfloat16* colK0, int lane) {
  FragU f;
  const int off = (lane < 16) ? 0 : 8;
  f.u[0] = *(const u32x4*)(colK0 + off);
  f.u[1] = *(const u32x4*)(colK0 + 16 + off);
  return f.m;
}
__device__ __forceinline__ v8f wmma_bf16(v16bf a, v16bf b, v8f c) {
  return __builtin_amdgcn_wmma_f32_16x16x32_bf16(false, a, false, b, (short)0, c,
                                                 false, false);
}
__device__ __forceinline__ int swiz(int n, int c) {  // bank-conflict-free 64x256 f32 tile
  return n * 256 + (c ^ (n & 63));
}

// ---- Tensor Data Mover: generic 2D bf16 tile DMA to LDS, rows padded to 80B pitch
// (64B data + 16B pad: 16B-aligned rows, bank-spread for b128 LDS reads).
__device__ __forceinline__ void tdm_load_tile2d(unsigned long long gaddr, unsigned ldsoff,
                                                unsigned tdim0, unsigned tdim1,
                                                unsigned stride0,
                                                unsigned tile0, unsigned tile1) {
  u32x4 g0;
  g0[0] = 1u;                                       // count=1, no gather/restore
  g0[1] = ldsoff;                                   // LDS byte address
  g0[2] = (unsigned)gaddr;                          // global addr [31:0]
  g0[3] = ((unsigned)(gaddr >> 32) & 0x01ffffffu) | (2u << 30);  // addr[56:32], type=2
  i32x8 g1;
  g1[0] = (int)((1u << 16)      // data_size = 2B
              | (1u << 20)      // pad_enable
              | (3u << 22)      // pad_interval: 16 dwords (one 64B row)
              | (3u << 25));    // pad_amount: 4 dwords (16B)
  g1[1] = (int)((tdim0 & 0xffffu) << 16);                 // tensor_dim0[15:0]
  g1[2] = (int)((tdim0 >> 16) | ((tdim1 & 0xffffu) << 16)); // dim0 hi, dim1 lo
  g1[3] = (int)((tdim1 >> 16) | (tile0 << 16));           // dim1 hi, tile_dim0
  g1[4] = (int)tile1;                                      // tile_dim1, tile_dim2=0
  g1[5] = (int)stride0;                                    // tensor_dim0_stride (elems)
  g1[6] = 0;
  g1[7] = 0;
  i32x4 gz = {0, 0, 0, 0};
#if defined(__clang_major__) && (__clang_major__ >= 23)
  i32x8 gz8 = {0, 0, 0, 0, 0, 0, 0, 0};
  __builtin_amdgcn_tensor_load_to_lds(g0, g1, gz, gz, gz8, 0);
#else
  __builtin_amdgcn_tensor_load_to_lds(g0, g1, gz, gz, 0);
#endif
}

#define VROWE 40                 /* padded LDS row pitch in bf16 elems (80B) */
#define VBUFB (CH * 80)          /* 20480 B per V buffer */
#define KBUFB (32 * 80)          /* 2560 B per K buffer  */
#define KOFFB (2 * VBUFB)        /* K buffers start at 40960 */

// ---------------- kernel 1: f32 -> bf16 convert (weights) ----------------
__global__ void sa_cvt_bf16(const float* __restrict__ s, __hip_bfloat16* __restrict__ d, int n) {
  int i = blockIdx.x * blockDim.x + threadIdx.x;
  if (i < n) d[i] = (__hip_bfloat16)s[i];
}

// ---------------- kernel 2: x[B,C,N] f32 -> xT[B,N,C] bf16 ----------------
__global__ __launch_bounds__(256) void sa_transpose_x(const float* __restrict__ x,
                                                      __hip_bfloat16* __restrict__ xT) {
  __shared__ float t[32][33];
  const int n0 = blockIdx.x * 32, c0 = blockIdx.y * 32, b = blockIdx.z;
  const int tx = threadIdx.x, ty = threadIdx.y;
  const float* xb = x + (size_t)b * CH * NPIX;
  __hip_bfloat16* xTb = xT + (size_t)b * NPIX * CH;
#pragma unroll
  for (int i = 0; i < 4; i++)
    t[ty + 8 * i][tx] = xb[(size_t)(c0 + ty + 8 * i) * NPIX + n0 + tx];
  __syncthreads();
#pragma unroll
  for (int i = 0; i < 4; i++)
    xTb[(size_t)(n0 + ty + 8 * i) * CH + c0 + tx] = (__hip_bfloat16)t[tx][ty + 8 * i];
}

// ---------------- kernel 3: projections (WMMA) ----------------
__global__ __launch_bounds__(256) void sa_proj(const __hip_bfloat16* __restrict__ xT,
                                               const __hip_bfloat16* __restrict__ wq,
                                               const __hip_bfloat16* __restrict__ wk,
                                               const __hip_bfloat16* __restrict__ wv,
                                               const float* __restrict__ bq,
                                               const float* __restrict__ bk,
                                               const float* __restrict__ bv,
                                               __hip_bfloat16* __restrict__ qT,
                                               __hip_bfloat16* __restrict__ kT,
                                               __hip_bfloat16* __restrict__ vv) {
  __shared__ float vlds[64 * 256];
  const int b = blockIdx.y, n0 = blockIdx.x * 64;
  const int tid = threadIdx.x, w = tid >> 5, lane = tid & 31, l16 = lane & 15;
  const int rs = w & 3, g = w >> 2;

  v8f accv[8], accp[2];
#pragma unroll
  for (int t = 0; t < 8; t++)
#pragma unroll
    for (int i = 0; i < 8; i++) accv[t][i] = 0.f;
#pragma unroll
  for (int t = 0; t < 2; t++)
#pragma unroll
    for (int i = 0; i < 8; i++) accp[t][i] = 0.f;

  const __hip_bfloat16* wp = (g == 0) ? wq : wk;
  const __hip_bfloat16* xrow = xT + (size_t)(b * NPIX + n0 + 16 * rs + l16) * CH;

  for (int c0 = 0; c0 < CH; c0 += 32) {
    v16bf a = load_fragA(xrow + c0, lane);
#pragma unroll
    for (int t = 0; t < 8; t++) {
      int c = g * 128 + t * 16 + l16;
      v16bf bf = load_fragB(wv + (size_t)c * CH + c0, lane);
      accv[t] = wmma_bf16(a, bf, accv[t]);
    }
#pragma unroll
    for (int t = 0; t < 2; t++) {
      int d = t * 16 + l16;
      v16bf bf = load_fragB(wp + (size_t)d * CH + c0, lane);
      accp[t] = wmma_bf16(a, bf, accp[t]);
    }
  }

  const float* biasp = (g == 0) ? bq : bk;
  __hip_bfloat16* outp = (g == 0) ? qT : kT;
  const int radd = (lane < 16) ? 0 : 8;
#pragma unroll
  for (int t = 0; t < 2; t++) {
    int d = t * 16 + l16;
    float bb = biasp[d];
#pragma unroll
    for (int r = 0; r < 8; r++) {
      int n = n0 + 16 * rs + r + radd;
      outp[(size_t)(b * NPIX + n) * DQK + d] = (__hip_bfloat16)(accp[t][r] + bb);
    }
  }
#pragma unroll
  for (int t = 0; t < 8; t++) {
    int c = g * 128 + t * 16 + l16;
    float bb = bv[c];
#pragma unroll
    for (int r = 0; r < 8; r++) {
      int nloc = 16 * rs + r + radd;
      vlds[swiz(nloc, c)] = accv[t][r] + bb;
    }
  }
  __syncthreads();
  for (int i = 0; i < 16; i++) {
    int c = i * 16 + (tid >> 4);
    int n4 = (tid & 15) * 4;
    union { __hip_bfloat16 h[4]; unsigned long long q; } pk;
#pragma unroll
    for (int j = 0; j < 4; j++) pk.h[j] = (__hip_bfloat16)vlds[swiz(n4 + j, c)];
    *(unsigned long long*)(vv + (size_t)(b * CH + c) * NPIX + n0 + n4) = pk.q;
  }
}

// ---------------- kernel 4: flash attention, 128-query tile ----------------
// 8 waves = 8 row-slices of 16 queries; each wave owns all 256 channels (16 acc tiles).
// No St duplication; TDM V tile amortized over 128 queries; TDM also stages the K tile.
__global__ __launch_bounds__(256) void sa_flash(const __hip_bfloat16* __restrict__ qT,
                                                const __hip_bfloat16* __restrict__ kT,
                                                const __hip_bfloat16* __restrict__ vv,
                                                const float* __restrict__ x,
                                                const float* __restrict__ gamma,
                                                float* __restrict__ out) {
  // loop phase: V bufs [0,40960) + K bufs [40960,46080); epilogue: 64KB O tile (aliased)
  __shared__ u32x4 smem4[65536 / 16];
  char* smem = (char*)smem4;
  float* olds = (float*)smem4;
  const unsigned lds_base = (unsigned)(uintptr_t)(void*)smem;

  const int b = blockIdx.y, n0 = blockIdx.x * 128;
  const int tid = threadIdx.x, w = tid >> 5, lane = tid & 31, l16 = lane & 15;
  const int radd = (lane < 16) ? 0 : 8;

  // Q as B-matrix of S^T = K x Q^T : lane = query column, K = d contiguous
  const __hip_bfloat16* qrow = qT + (size_t)(b * NPIX + n0 + 16 * w + l16) * DQK;
  const v16bf qB = load_fragB(qrow, lane);

  v8f acc[16];
#pragma unroll
  for (int t = 0; t < 16; t++)
#pragma unroll
    for (int i = 0; i < 8; i++) acc[t][i] = 0.f;
  float m_run = -1e30f, l_run = 0.f;

  const unsigned long long k_addr = (unsigned long long)(uintptr_t)(kT + (size_t)b * NPIX * DQK);
  const unsigned long long v_addr = (unsigned long long)(uintptr_t)(vv + (size_t)(b * CH) * NPIX);

  if (w == 0) {  // prefetch tile 0 (V then K)
    tdm_load_tile2d(v_addr, lds_base, NPIX, CH, NPIX, 32, CH);
    tdm_load_tile2d(k_addr, lds_base + KOFFB, 32, NPIX, 32, 32, 32);
  }

  const int NIT = NPIX / 32;
  for (int it = 0; it < NIT; ++it) {
    const int kb = it * 32;
    if (w == 0) __builtin_amdgcn_s_wait_tensorcnt((short)0);   // buf[it&1] landed
    __syncthreads();   // publish; all waves done reading buf[(it+1)&1] from it-1
    if (w == 0 && it + 1 < NIT) {
      const int nb = (it + 1) & 1;
      tdm_load_tile2d(v_addr + (unsigned long long)(kb + 32) * 2,
                      lds_base + (unsigned)nb * VBUFB, NPIX, CH, NPIX, 32, CH);
      tdm_load_tile2d(k_addr + (unsigned long long)(kb + 32) * DQK * 2,
                      lds_base + KOFFB + (unsigned)nb * KBUFB, 32, NPIX, 32, 32, 32);
    }

    // S^T tiles from the LDS-staged K tile: lane = query, VGPR = key
    const __hip_bfloat16* kvt =
        (const __hip_bfloat16*)(smem + KOFFB + (it & 1) * KBUFB);
    v16bf a0 = load_fragA(kvt + (size_t)l16 * VROWE, lane);
    v16bf a1 = load_fragA(kvt + (size_t)(16 + l16) * VROWE, lane);
    v8f z;
#pragma unroll
    for (int i = 0; i < 8; i++) z[i] = 0.f;
    v8f s0 = wmma_bf16(a0, qB, z);
    v8f s1 = wmma_bf16(a1, qB, z);

    // online softmax (lane = query; pair lane^16 holds the other 16 keys)
    float tm = s0[0];
#pragma unroll
    for (int i = 1; i < 8; i++) tm = fmaxf(tm, s0[i]);
#pragma unroll
    for (int i = 0; i < 8; i++) tm = fmaxf(tm, s1[i]);
    tm = fmaxf(tm, __shfl_xor(tm, 16, 32));
    float mn = fmaxf(m_run, tm);
    float corr = __expf(m_run - mn);
    float p0[8], p1[8], ps = 0.f;
#pragma unroll
    for (int i = 0; i < 8; i++) { p0[i] = __expf(s0[i] - mn); ps += p0[i]; }
#pragma unroll
    for (int i = 0; i < 8; i++) { p1[i] = __expf(s1[i] - mn); ps += p1[i]; }
    ps += __shfl_xor(ps, 16, 32);
    l_run = l_run * corr + ps;
    m_run = mn;

#pragma unroll
    for (int r = 0; r < 8; r++) {
      float sr = __shfl(corr, r + radd, 32);
#pragma unroll
      for (int t = 0; t < 16; t++) acc[t][r] *= sr;
    }

    // P (A-fragment): C-layout of S^T == A-layout of P, zero cross-lane movement
    v16bf pa;
#pragma unroll
    for (int i = 0; i < 8; i++) pa[i] = (__bf16)p0[i];
#pragma unroll
    for (int i = 0; i < 8; i++) pa[8 + i] = (__bf16)p1[i];

    // O += P @ V from the TDM-staged LDS tile (80B row pitch), all 256 channels
    const __hip_bfloat16* vrow = (const __hip_bfloat16*)(smem + (it & 1) * VBUFB);
#pragma unroll
    for (int t = 0; t < 16; t++) {
      int c = t * 16 + l16;
      v16bf vB = load_fragB(vrow + (size_t)c * VROWE, lane);
      acc[t] = wmma_bf16(pa, vB, acc[t]);
    }
  }

  __syncthreads();  // retire V/K buffers before reusing LDS as the O tile

  float linv[8];
#pragma unroll
  for (int r = 0; r < 8; r++) linv[r] = 1.0f / __shfl(l_run, r + radd, 32);
  const float gm = gamma[0];

  // two 64-row passes through the 64KB swizzled O tile
  for (int pass = 0; pass < 2; pass++) {
    if ((w >> 2) == pass) {
#pragma unroll
      for (int t = 0; t < 16; t++) {
        int c = t * 16 + l16;
#pragma unroll
        for (int r = 0; r < 8; r++) {
          int nloc = 16 * (w & 3) + r + radd;
          olds[swiz(nloc, c)] = acc[t][r] * linv[r];
        }
      }
    }
    __syncthreads();
    for (int i = 0; i < 16; i++) {
      int c = i * 16 + (tid >> 4);
      int n4 = (tid & 15) * 4;
      size_t gi = (size_t)(b * CH + c) * NPIX + n0 + pass * 64 + n4;
      const float4 xv = *(const float4*)(x + gi);
      float4 ov;
      ov.x = gm * olds[swiz(n4 + 0, c)] + xv.x;
      ov.y = gm * olds[swiz(n4 + 1, c)] + xv.y;
      ov.z = gm * olds[swiz(n4 + 2, c)] + xv.z;
      ov.w = gm * olds[swiz(n4 + 3, c)] + xv.w;
      *(float4*)(out + gi) = ov;
    }
    __syncthreads();
  }
}

extern "C" void kernel_launch(void* const* d_in, const int* in_sizes, int n_in,
                              void* d_out, int out_size, void* d_ws, size_t ws_size,
                              hipStream_t stream) {
  const float* x     = (const float*)d_in[0];
  const float* Wq    = (const float*)d_in[1];
  const float* bq    = (const float*)d_in[2];
  const float* Wk    = (const float*)d_in[3];
  const float* bk    = (const float*)d_in[4];
  const float* Wv    = (const float*)d_in[5];
  const float* bv    = (const float*)d_in[6];
  const float* gamma = (const float*)d_in[7];
  float* out = (float*)d_out;

  char* ws = (char*)d_ws;
  size_t off = 0;
  __hip_bfloat16* wq_bf = (__hip_bfloat16*)(ws + off); off += (size_t)DQK * CH * 2;
  __hip_bfloat16* wk_bf = (__hip_bfloat16*)(ws + off); off += (size_t)DQK * CH * 2;
  __hip_bfloat16* wv_bf = (__hip_bfloat16*)(ws + off); off += (size_t)CH * CH * 2;
  __hip_bfloat16* xT    = (__hip_bfloat16*)(ws + off); off += (size_t)BB * NPIX * CH * 2;
  __hip_bfloat16* qT    = (__hip_bfloat16*)(ws + off); off += (size_t)BB * NPIX * DQK * 2;
  __hip_bfloat16* kTp   = (__hip_bfloat16*)(ws + off); off += (size_t)BB * NPIX * DQK * 2;
  __hip_bfloat16* vv    = (__hip_bfloat16*)(ws + off); off += (size_t)BB * CH * NPIX * 2;
  (void)ws_size; (void)in_sizes; (void)n_in; (void)out_size;

  sa_cvt_bf16<<<(DQK * CH + 255) / 256, 256, 0, stream>>>(Wq, wq_bf, DQK * CH);
  sa_cvt_bf16<<<(DQK * CH + 255) / 256, 256, 0, stream>>>(Wk, wk_bf, DQK * CH);
  sa_cvt_bf16<<<(CH * CH + 255) / 256, 256, 0, stream>>>(Wv, wv_bf, CH * CH);

  sa_transpose_x<<<dim3(NPIX / 32, CH / 32, BB), dim3(32, 8), 0, stream>>>(x, xT);

  sa_proj<<<dim3(NPIX / 64, BB), 256, 0, stream>>>(xT, wq_bf, wk_bf, wv_bf,
                                                   bq, bk, bv, qT, kTp, vv);

  sa_flash<<<dim3(NPIX / 128, BB), 256, 0, stream>>>(qT, kTp, vv, x, gamma, out);
}
